// Model_54288386621933
// MI455X (gfx1250) — compile-verified, matrix operand order checked by reading, not run on hardware
//
#include <hip/hip_runtime.h>

// ---------------------------------------------------------------------------
// CDNA5 (gfx1250, wave32) implementation of the TimeXer-style encoder.
// All dense GEMMs run through a v_wmma_f32_16x16x32_f16 tiled kernel
// (f32 activations converted to f16 on the fly, weights pre-converted).
// K is guaranteed to be a multiple of 32 for every call (embed is padded),
// so the LDS staging path is branch-free and uses b128 loads/stores.
// ---------------------------------------------------------------------------

typedef __attribute__((ext_vector_type(16))) _Float16 v16h;
typedef __attribute__((ext_vector_type(8)))  float    v8f;

#define B_    64
#define NV_   32
#define SEQ_  512
#define SEG_  32
#define SN_   33            // SEG+1
#define DM_   512
#define BN_   2048          // B*NV
#define TOKS_ 67584         // BN*SN

// ---------------------------------------------------------------------------
// f32 -> f16 weight conversion
// ---------------------------------------------------------------------------
__global__ void k_cvt_f16(const float* __restrict__ s, _Float16* __restrict__ d, int n) {
  int i = blockIdx.x * 256 + threadIdx.x;
  if (i < n) d[i] = (_Float16)s[i];
}

// f32 (N x Ks) -> f16 (N x Kd) with zero padding of the K tail.
__global__ void k_cvt_pad(const float* __restrict__ s, _Float16* __restrict__ d,
                          int Ks, int Kd, int total) {
  int i = blockIdx.x * 256 + threadIdx.x;
  if (i >= total) return;
  const int n = i / Kd, k = i % Kd;
  d[i] = (k < Ks) ? (_Float16)s[(size_t)n * Ks + k] : (_Float16)0.f;
}

// ---------------------------------------------------------------------------
// Generic GEMM: C[M,N] = act(A[M,K] @ W[N,K]^T + bias), f16 WMMA, f32 acc.
// grid = (M/64, ceil(N/64)); block = 256 (8 waves).
// Requirements: M % 64 == 0, K % 32 == 0, A/W/lda/K 16-byte aligned.
// ---------------------------------------------------------------------------
__global__ __launch_bounds__(256)
void k_gemm(const float* __restrict__ A, int lda,
            const _Float16* __restrict__ W,
            const float* __restrict__ bias,
            float* __restrict__ C, int ldc,
            int N, int K, int act) {
  __shared__ __align__(16) _Float16 As[64][40];
  __shared__ __align__(16) _Float16 Bs[64][40];
  const int m0 = blockIdx.x * 64;
  const int n0 = blockIdx.y * 64;
  const int tid  = threadIdx.x;
  const int wave = tid >> 5;
  const int lane = tid & 31;
  const int hl = lane & 15;            // half-lane (row/col within 16)
  const int hi = lane >> 4;            // 0 / 1
  const int mrow = (wave >> 1) * 16;   // wave's M sub-tile
  const int ncol = (wave & 1) * 32;    // wave's N sub-tile pair
  const int lr = tid >> 2;             // load row 0..63
  const int lc = (tid & 3) * 8;        // load col start {0,8,16,24}

  v8f acc0 = {};
  v8f acc1 = {};

  const float*    Arow = A + (size_t)(m0 + lr) * lda + lc;
  const int       wr   = n0 + lr;
  const _Float16* Wrow = W + (size_t)wr * K + lc;

  for (int kk = 0; kk < K; kk += 32) {
    // ---- stage A (f32 -> f16), 8 elems per thread, b128 traffic ----
    const float4* ap = (const float4*)(Arow + kk);
    float4 a0 = ap[0];
    float4 a1 = ap[1];
    union { _Float16 h[8]; uint4 u; } pk;
    pk.h[0] = (_Float16)a0.x; pk.h[1] = (_Float16)a0.y;
    pk.h[2] = (_Float16)a0.z; pk.h[3] = (_Float16)a0.w;
    pk.h[4] = (_Float16)a1.x; pk.h[5] = (_Float16)a1.y;
    pk.h[6] = (_Float16)a1.z; pk.h[7] = (_Float16)a1.w;
    *(uint4*)&As[lr][lc] = pk.u;
    if (kk + 32 < K) __builtin_prefetch((const void*)(ap + 8), 0, 1);

    // ---- stage W (already f16), one b128 per thread ----
    uint4 wv = make_uint4(0u, 0u, 0u, 0u);
    if (wr < N) wv = *(const uint4*)(Wrow + kk);
    *(uint4*)&Bs[lr][lc] = wv;
    __syncthreads();

    // ---- gather fragments per CDNA5 ISA 16-bit A (16x32) / B (32x16) layouts
    v16h af, bf0, bf1;
#pragma unroll
    for (int i = 0; i < 16; ++i) {
      const int j  = i >> 1;
      const int kx = (hi ? 8 : 0) + ((j & 4) ? 16 : 0) + 2 * (j & 3) + (i & 1);
      af[i]  = As[mrow + hl][kx];
      bf0[i] = Bs[ncol + hl][kx];
      bf1[i] = Bs[ncol + 16 + hl][kx];
    }
    acc0 = __builtin_amdgcn_wmma_f32_16x16x32_f16(false, af, false, bf0,
                                                  (short)0, acc0, false, false);
    acc1 = __builtin_amdgcn_wmma_f32_16x16x32_f16(false, af, false, bf1,
                                                  (short)0, acc1, false, false);
    __syncthreads();
  }

  const int col0 = n0 + ncol + hl;
  const int col1 = col0 + 16;
  const float bv0 = (bias && col0 < N) ? bias[col0] : 0.f;
  const float bv1 = (bias && col1 < N) ? bias[col1] : 0.f;
#pragma unroll
  for (int v = 0; v < 8; ++v) {
    const int row = m0 + mrow + hi * 8 + v;
    if (col0 < N) {
      float x = acc0[v] + bv0;
      if (act == 1) x = fmaxf(x, 0.f);
      else if (act == 2) x = 0.5f * x * (1.f + erff(x * 0.70710678118f));
      C[(size_t)row * ldc + col0] = x;
    }
    if (col1 < N) {
      float x = acc1[v] + bv1;
      if (act == 1) x = fmaxf(x, 0.f);
      else if (act == 2) x = 0.5f * x * (1.f + erff(x * 0.70710678118f));
      C[(size_t)row * ldc + col1] = x;
    }
  }
}

// ---------------------------------------------------------------------------
// Instance normalization: mean/std over SEQ per (b,n); writes xT[bn][s].
// grid = BN, block = 256.
// ---------------------------------------------------------------------------
__global__ void k_norm(const float* __restrict__ x_enc, float* __restrict__ xT,
                       float* __restrict__ meanb, float* __restrict__ stdb) {
  const int bn = blockIdx.x;
  const int b = bn >> 5, n = bn & 31;
  const int t = threadIdx.x;
  const float* src = x_enc + (size_t)b * SEQ_ * NV_ + n;
  float a0 = src[(size_t)t * NV_];
  float a1 = src[(size_t)(t + 256) * NV_];
  __shared__ float red[256];
  red[t] = a0 + a1;
  __syncthreads();
  for (int o = 128; o > 0; o >>= 1) { if (t < o) red[t] += red[t + o]; __syncthreads(); }
  const float m = red[0] * (1.f / SEQ_);
  __syncthreads();
  float d0 = a0 - m, d1 = a1 - m;
  red[t] = d0 * d0 + d1 * d1;
  __syncthreads();
  for (int o = 128; o > 0; o >>= 1) { if (t < o) red[t] += red[t + o]; __syncthreads(); }
  const float sd = sqrtf(red[0] * (1.f / SEQ_) + 1e-5f);
  const float inv = 1.f / sd;
  xT[(size_t)bn * SEQ_ + t]       = d0 * inv;
  xT[(size_t)bn * SEQ_ + t + 256] = d1 * inv;
  if (t == 0) { meanb[bn] = m; stdb[bn] = sd; }
}

// ---------------------------------------------------------------------------
// 1D conv (kernel 17, pad 8, cross-correlation) + residual, scattered into a
// K=32 zero-padded patch matrix: convx32[(bn*SEG+seg)][p] (p<16 data, else 0).
// ---------------------------------------------------------------------------
__global__ void k_conv(const float* __restrict__ xT, const float* __restrict__ ce,
                       float* __restrict__ convx32, int total) {
  int idx = blockIdx.x * 256 + threadIdx.x;
  if (idx >= total) return;
  const int bn = idx >> 9, s = idx & 511;
  const float* xr = xT + (size_t)bn * SEQ_;
  float acc = xr[s];
#pragma unroll
  for (int j = 0; j < 17; ++j) {
    int p = s - 8 + j;
    if (p >= 0 && p < SEQ_) acc += ce[j] * xr[p];
  }
  const int seg = s >> 4, p = s & 15;
  const size_t row = (size_t)bn * SEG_ + seg;
  convx32[row * 32 + p]      = acc;
  convx32[row * 32 + 16 + p] = 0.f;
}

// ---------------------------------------------------------------------------
// Assemble tokens: embed + sinusoidal PE for s<32, glb token for s==32.
// ---------------------------------------------------------------------------
__global__ void k_embed_finish(const float* __restrict__ embt, const float* __restrict__ glb,
                               float* __restrict__ tok, int total) {
  int idx = blockIdx.x * 256 + threadIdx.x;
  if (idx >= total) return;
  const int d = idx & 511;
  const int rs = idx >> 9;
  const int s = rs % SN_;
  const int bn = rs / SN_;
  float v;
  if (s == 32) {
    v = glb[(size_t)(bn & 31) * DM_ + d];
  } else {
    const int i = d >> 1;
    const float freq = expf(-(float)(2 * i) * 0.0179889450f);  // ln(1e4)/512
    const float ang = (float)s * freq;
    const float pe = (d & 1) ? cosf(ang) : sinf(ang);
    v = embt[((size_t)bn * SEG_ + s) * DM_ + d] + pe;
  }
  tok[idx] = v;
}

// ---------------------------------------------------------------------------
// Attention for S=33, DH=64. grid = (BN, NH), block = 64. Writes o in-place
// into the q buffer (same (bn,s,h,d) layout).
// ---------------------------------------------------------------------------
__global__ void k_attn(float* __restrict__ q, const float* __restrict__ k,
                       const float* __restrict__ v) {
  const int bn = blockIdx.x, h = blockIdx.y;
  __shared__ float qs[SN_][64], ks[SN_][64], vs[SN_][64];
  __shared__ float sc[SN_], aw[SN_], invs;
  const int t = threadIdx.x;
  const size_t base = ((size_t)bn * SN_) * DM_ + h * 64;
  for (int i = t; i < SN_ * 64; i += 64) {
    const int s = i >> 6, d = i & 63;
    qs[s][d] = q[base + (size_t)s * DM_ + d];
    ks[s][d] = k[base + (size_t)s * DM_ + d];
    vs[s][d] = v[base + (size_t)s * DM_ + d];
  }
  __syncthreads();
  for (int l = 0; l < SN_; ++l) {
    if (t < SN_) {
      float a = 0.f;
      for (int d = 0; d < 64; ++d) a += qs[l][d] * ks[t][d];
      sc[t] = a * 0.125f;
    }
    __syncthreads();
    if (t == 0) {
      float mx = sc[0];
      for (int s = 1; s < SN_; ++s) mx = fmaxf(mx, sc[s]);
      float sum = 0.f;
      for (int s = 0; s < SN_; ++s) { float e = expf(sc[s] - mx); aw[s] = e; sum += e; }
      invs = 1.f / sum;
    }
    __syncthreads();
    float a = 0.f;
    for (int s = 0; s < SN_; ++s) a += aw[s] * vs[s][t];
    q[base + (size_t)l * DM_ + t] = a * invs;
    __syncthreads();
  }
}

// ---------------------------------------------------------------------------
// LayerNorm over 512 with optional residual: x = LN(x + y)*g + b (in place).
// grid = rows, block = 256. y may be null.
// ---------------------------------------------------------------------------
__global__ void k_ln(float* __restrict__ x, int xstride,
                     const float* __restrict__ y, int ystride,
                     const float* __restrict__ g, const float* __restrict__ bb) {
  const int row = blockIdx.x, t = threadIdx.x;
  float* xr = x + (size_t)row * xstride;
  const float* yr = y ? y + (size_t)row * ystride : nullptr;
  float v0 = xr[t]       + (yr ? yr[t]       : 0.f);
  float v1 = xr[t + 256] + (yr ? yr[t + 256] : 0.f);
  __shared__ float red[256];
  red[t] = v0 + v1;
  __syncthreads();
  for (int o = 128; o > 0; o >>= 1) { if (t < o) red[t] += red[t + o]; __syncthreads(); }
  const float m = red[0] * (1.f / DM_);
  __syncthreads();
  float d0 = v0 - m, d1 = v1 - m;
  red[t] = d0 * d0 + d1 * d1;
  __syncthreads();
  for (int o = 128; o > 0; o >>= 1) { if (t < o) red[t] += red[t + o]; __syncthreads(); }
  const float inv = rsqrtf(red[0] * (1.f / DM_) + 1e-5f);
  xr[t]       = d0 * inv * g[t]       + bb[t];
  xr[t + 256] = d1 * inv * g[t + 256] + bb[t + 256];
}

// ---------------------------------------------------------------------------
// core[bn,h,c] = gelu(sum_e mh[bn,h*64+e]*w3[h,c,e] + b3[h,c]); grid=BN, blk=64.
// ---------------------------------------------------------------------------
__global__ void k_core(const float* __restrict__ mh, const float* __restrict__ w3,
                       const float* __restrict__ b3, float* __restrict__ core) {
  const int bn = blockIdx.x, t = threadIdx.x;
  const int h = t >> 3, c = t & 7;
  const float* m = mh + (size_t)bn * DM_ + h * 64;
  const float* w = w3 + (size_t)h * 512 + c * 64;
  float acc = b3[h * 8 + c];
  for (int e = 0; e < 64; ++e) acc += m[e] * w[e];
  core[(size_t)bn * 64 + t] = 0.5f * acc * (1.f + erff(acc * 0.70710678118f));
}

// ---------------------------------------------------------------------------
// cmsum[b,c] = sum_n cm[b,n,c]*softmax_n(cm[b,n,c]); grid=B, block=64.
// ---------------------------------------------------------------------------
__global__ void k_cmagg(const float* __restrict__ cm, float* __restrict__ cmsum) {
  const int b = blockIdx.x, c = threadIdx.x;
  float vals[NV_];
  float mx = -3.4e38f;
  for (int n = 0; n < NV_; ++n) {
    float v = cm[((size_t)(b * NV_ + n)) * 64 + c];
    vals[n] = v;
    mx = fmaxf(mx, v);
  }
  float sum = 0.f;
  for (int n = 0; n < NV_; ++n) sum += expf(vals[n] - mx);
  const float inv = 1.f / sum;
  float out = 0.f;
  for (int n = 0; n < NV_; ++n) out += vals[n] * expf(vals[n] - mx) * inv;
  cmsum[b * 64 + c] = out;
}

// cat[bn, 0:512]=tok[bn,32,:], cat[bn,512:576]=cmsum[b].
__global__ void k_cat(const float* __restrict__ tok, const float* __restrict__ cmsum,
                      float* __restrict__ cat, int total) {
  int idx = blockIdx.x * 256 + threadIdx.x;
  if (idx >= total) return;
  const int bn = idx / 576, j = idx % 576;
  cat[idx] = (j < DM_) ? tok[((size_t)bn * SN_ + 32) * DM_ + j]
                       : cmsum[(bn >> 5) * 64 + (j - DM_)];
}

// xh[bn, d*33+s] = tok[bn, s, d]
__global__ void k_repack(const float* __restrict__ tok, float* __restrict__ xh, int total) {
  int idx = blockIdx.x * 256 + threadIdx.x;
  if (idx >= total) return;
  const int d = idx & 511;
  const int rs = idx >> 9;
  const int s = rs % SN_;
  const int bn = rs / SN_;
  xh[(size_t)bn * (DM_ * SN_) + d * SN_ + s] = tok[idx];
}

// out[b,p,n] = head[bn,p]*std[bn] + mean[bn]
__global__ void k_out(const float* __restrict__ head, const float* __restrict__ stdb,
                      const float* __restrict__ meanb, float* __restrict__ out, int total) {
  int idx = blockIdx.x * 256 + threadIdx.x;
  if (idx >= total) return;
  const int n = idx & 31;
  const int rest = idx >> 5;
  const int p = rest % 96;
  const int b = rest / 96;
  const int bn = b * NV_ + n;
  out[idx] = head[(size_t)bn * 96 + p] * stdb[bn] + meanb[bn];
}

// ---------------------------------------------------------------------------
// Host orchestration
// ---------------------------------------------------------------------------
extern "C" void kernel_launch(void* const* d_in, const int* in_sizes, int n_in,
                              void* d_out, int out_size, void* d_ws, size_t ws_size,
                              hipStream_t stream) {
  (void)in_sizes; (void)n_in; (void)out_size; (void)ws_size;
  // jax tree-flatten order (sorted dict keys; layers list in order)
  const float* b_head   = (const float*)d_in[0];
  const float* conv_emb = (const float*)d_in[1];
  const float* glb      = (const float*)d_in[2];
  const float* L[2][28];
  for (int i = 0; i < 28; ++i) { L[0][i] = (const float*)d_in[3 + i]; L[1][i] = (const float*)d_in[31 + i]; }
  const float* norm_b = (const float*)d_in[59];
  const float* norm_g = (const float*)d_in[60];
  const float* w_head = (const float*)d_in[61];
  const float* w_val  = (const float*)d_in[62];
  const float* x_enc  = (const float*)d_in[63];
  float* out = (float*)d_out;

  enum { IB2=0, IB3, IB4, IB5, IB6, IBK, IBO, IBQ, IBV, IC1, IC2, ICB1, ICB2,
         IG1, IG2, IG3, IN1, IN2, IN3, IW2, IW3, IW4, IW5, IW6, IWK, IWO, IWQ, IWV };

  // ---- workspace allocator (16-byte aligned, units of floats) ----
  float* ws = (float*)d_ws;
  size_t off = 0;
  auto allocf = [&](size_t n) { float* p = ws + off; off += (n + 3) & ~(size_t)3; return p; };
  auto alloch = [&](size_t n) { _Float16* p = (_Float16*)(ws + off);
                                off += (((n + 1) / 2) + 3) & ~(size_t)3; return p; };

  float* meanb   = allocf(BN_);
  float* stdb    = allocf(BN_);
  float* xT      = allocf((size_t)BN_ * SEQ_);
  float* convx32 = allocf((size_t)BN_ * SEG_ * 32);
  float* tok     = allocf((size_t)TOKS_ * DM_);
  float* qbuf    = allocf((size_t)TOKS_ * DM_);   // also embt / t5 / xh
  float* kbuf    = allocf((size_t)TOKS_ * DM_);   // also y / t6
  float* vbuf    = allocf((size_t)TOKS_ * DM_);   // also mh / cat / hid
  float* core    = allocf((size_t)BN_ * 64);
  float* cmb     = allocf((size_t)BN_ * 64);
  float* cmsum   = allocf((size_t)B_ * 64);
  float* headv   = allocf((size_t)BN_ * 96);

  auto CVT = [&](const float* s, _Float16* d, int n) {
    k_cvt_f16<<<(n + 255) / 256, 256, 0, stream>>>(s, d, n);
  };
  auto GEMM = [&](const float* A, int lda, const _Float16* W, const float* bias,
                  float* C, int ldc, int M, int N, int K, int act) {
    dim3 g(M / 64, (N + 63) / 64);
    k_gemm<<<g, 256, 0, stream>>>(A, lda, W, bias, C, ldc, N, K, act);
  };

  // ---- convert weights once per launch ----
  _Float16 *wqh[2], *wkh[2], *wvh[2], *woh[2], *w2h[2], *w4h[2], *w5h[2], *w6h[2], *c1h[2], *c2h[2];
  for (int l = 0; l < 2; ++l) {
    wqh[l] = alloch(512 * 512); CVT(L[l][IWQ], wqh[l], 512 * 512);
    wkh[l] = alloch(512 * 512); CVT(L[l][IWK], wkh[l], 512 * 512);
    wvh[l] = alloch(512 * 512); CVT(L[l][IWV], wvh[l], 512 * 512);
    woh[l] = alloch(512 * 512); CVT(L[l][IWO], woh[l], 512 * 512);
    w2h[l] = alloch(512 * 512); CVT(L[l][IW2], w2h[l], 512 * 512);
    w4h[l] = alloch(64 * 64);   CVT(L[l][IW4], w4h[l], 64 * 64);
    w5h[l] = alloch(512 * 576); CVT(L[l][IW5], w5h[l], 512 * 576);
    w6h[l] = alloch(512 * 512); CVT(L[l][IW6], w6h[l], 512 * 512);
    c1h[l] = alloch(2048 * 512); CVT(L[l][IC1], c1h[l], 2048 * 512);
    c2h[l] = alloch(512 * 2048); CVT(L[l][IC2], c2h[l], 512 * 2048);
  }
  _Float16* wvalh  = alloch(512 * 32);  // zero-padded K: 16 -> 32
  k_cvt_pad<<<(512 * 32 + 255) / 256, 256, 0, stream>>>(w_val, wvalh, 16, 32, 512 * 32);
  _Float16* wheadh = alloch(96 * 16896);
  CVT(w_head, wheadh, 96 * 16896);

  // ---- front end ----
  k_norm<<<BN_, 256, 0, stream>>>(x_enc, xT, meanb, stdb);
  k_conv<<<(BN_ * SEQ_) / 256, 256, 0, stream>>>(xT, conv_emb, convx32, BN_ * SEQ_);
  // patch embed: (BN*SEG,32pad) @ w_val^T -> qbuf (embt)
  GEMM(convx32, 32, wvalh, nullptr, qbuf, DM_, BN_ * SEG_, DM_, 32, 0);
  k_embed_finish<<<((size_t)TOKS_ * DM_) / 256, 256, 0, stream>>>(qbuf, glb, tok, TOKS_ * DM_);

  // ---- encoder layers ----
  for (int l = 0; l < 2; ++l) {
    // QKV
    GEMM(tok, DM_, wqh[l], L[l][IBQ], qbuf, DM_, TOKS_, DM_, DM_, 0);
    GEMM(tok, DM_, wkh[l], L[l][IBK], kbuf, DM_, TOKS_, DM_, DM_, 0);
    GEMM(tok, DM_, wvh[l], L[l][IBV], vbuf, DM_, TOKS_, DM_, DM_, 0);
    // attention (o written into qbuf)
    k_attn<<<dim3(BN_, 8), 64, 0, stream>>>(qbuf, kbuf, vbuf);
    // o @ wo^T + bo -> kbuf, then LN1 residual
    GEMM(qbuf, DM_, woh[l], L[l][IBO], kbuf, DM_, TOKS_, DM_, DM_, 0);
    k_ln<<<TOKS_, 256, 0, stream>>>(tok, DM_, kbuf, DM_, L[l][IG1], L[l][IN1]);

    // global-token cross-variable block
    const float* xg = tok + 32 * DM_;                 // rows strided by 33*512
    GEMM(xg, SN_ * DM_, w2h[l], L[l][IB2], vbuf, DM_, BN_, DM_, DM_, 2);     // mh (gelu)
    k_core<<<BN_, 64, 0, stream>>>(vbuf, L[l][IW3], L[l][IB3], core);
    GEMM(core, 64, w4h[l], L[l][IB4], cmb, 64, BN_, 64, 64, 0);              // cm
    k_cmagg<<<B_, 64, 0, stream>>>(cmb, cmsum);
    k_cat<<<(BN_ * 576 + 255) / 256, 256, 0, stream>>>(tok, cmsum, vbuf, BN_ * 576);
    GEMM(vbuf, 576, w5h[l], L[l][IB5], qbuf, DM_, BN_, DM_, 576, 2);         // t5 (gelu)
    GEMM(qbuf, DM_, w6h[l], L[l][IB6], kbuf, DM_, BN_, DM_, DM_, 0);         // t6
    k_ln<<<BN_, 256, 0, stream>>>(tok + 32 * DM_, SN_ * DM_, kbuf, DM_, L[l][IG2], L[l][IN2]);

    // FFN, chunked over rows to bound the hidden buffer
    const int CH = 8448;                              // 67584 / 8
    for (int c0 = 0; c0 < TOKS_; c0 += CH) {
      GEMM(tok + (size_t)c0 * DM_, DM_, c1h[l], L[l][ICB1], vbuf, 2048, CH, 2048, DM_, 1);
      GEMM(vbuf, 2048, c2h[l], L[l][ICB2], kbuf + (size_t)c0 * DM_, DM_, CH, DM_, 2048, 0);
    }
    k_ln<<<TOKS_, 256, 0, stream>>>(tok, DM_, kbuf, DM_, L[l][IG3], L[l][IN3]);
  }

  // ---- final LN, head, de-normalization ----
  k_ln<<<TOKS_, 256, 0, stream>>>(tok, DM_, nullptr, 0, norm_g, norm_b);
  k_repack<<<((size_t)TOKS_ * DM_) / 256, 256, 0, stream>>>(tok, qbuf, TOKS_ * DM_);
  GEMM(qbuf, DM_ * SN_, wheadh, b_head, headv, 96, BN_, 96, DM_ * SN_, 0);
  k_out<<<(B_ * 96 * NV_) / 256, 256, 0, stream>>>(headv, stdb, meanb, out, B_ * 96 * NV_);
}
